// Model_54511724920997
// MI455X (gfx1250) — matrix operand order repaired and checked by reading, not compiled, not run on hardware
//
#include <hip/hip_runtime.h>
#include <hip/hip_bf16.h>

typedef __attribute__((ext_vector_type(16))) __bf16 v16bf;
typedef __attribute__((ext_vector_type(8)))  __bf16 v8bf;
typedef __attribute__((ext_vector_type(8)))  float  v8f;

#define N_OP  50000
#define N_DEV 16
#define E_PP  800000
#define E_PD  800000
#define E_DD  240

__device__ __forceinline__ float elu1(float x) {
  return x > 0.f ? x : (__expf(x) - 1.f);
}

__device__ __forceinline__ unsigned short bfbits(float f) {
  unsigned int u = __builtin_bit_cast(unsigned int, f);
  unsigned int r = u + 0x7FFFu + ((u >> 16) & 1u);  // round-to-nearest-even
  return (unsigned short)(r >> 16);
}

__device__ __forceinline__ __bf16 f2bf(float f) {
  unsigned short h = bfbits(f);
  return __builtin_bit_cast(__bf16, h);
}

// ---------------------------------------------------------------------------
// Packed f32 -> bf16 conversion (2 elements / thread).
// ---------------------------------------------------------------------------
__global__ void cvt_bf16_kernel(const float* __restrict__ src, __bf16* __restrict__ dst,
                                int npairs) {
  const int i = blockIdx.x * blockDim.x + threadIdx.x;
  if (i >= npairs) return;
  const float2 f = ((const float2*)src)[i];
  ushort2 h;
  h.x = bfbits(f.x);
  h.y = bfbits(f.y);
  ((ushort2*)dst)[i] = h;
}

// ---------------------------------------------------------------------------
// Pack W[K x 64] (f32, row-major) into per-lane WMMA B fragments (bf16):
//   Bfrag[((kblk*4 + ntile)*32 + lane)*16 + i] = W[kb][n]
//   kb = kblk*32 + (lane>>4)*16 + i ; n = ntile*16 + (lane&15)
// ---------------------------------------------------------------------------
__global__ void pack_w_kernel(const float* __restrict__ W, __bf16* __restrict__ Bfrag,
                              int K) {
  const int idx = blockIdx.x * blockDim.x + threadIdx.x;
  const int total = (K / 32) * 2048;
  if (idx >= total) return;
  const int i     = idx & 15;
  const int lane  = (idx >> 4) & 31;
  const int ntile = (idx >> 9) & 3;
  const int kblk  = idx >> 11;
  const int n  = ntile * 16 + (lane & 15);
  const int kb = kblk * 32 + (lane >> 4) * 16 + i;
  Bfrag[idx] = f2bf(W[(size_t)kb * 64 + n]);
}

// ---------------------------------------------------------------------------
// C[M x 64] = act(A[M x K](bf16) @ W(packed frags) + bias)
// via v_wmma_f32_16x16x32_bf16. One wave per 16x16 tile, 4 waves cover N=64.
// K is a compile-time constant -> fully unrolled (K/32 static WMMAs).
// Inner loop is pure vector loads + WMMA (no conversion VALU).
// ---------------------------------------------------------------------------
template <int K>
__global__ void __launch_bounds__(128)
gemm64_bf_kernel(const __bf16* __restrict__ A, const __bf16* __restrict__ Bfrag,
                 const float* __restrict__ bias, float* __restrict__ C, int act) {
  const int lane = threadIdx.x & 31;
  const int wave = threadIdx.x >> 5;     // N tile (0..3)
  const int l15  = lane & 15;
  const int hi   = lane >> 4;            // half-wave selector
  const int row  = blockIdx.x * 16 + l15;
  const int n    = wave * 16 + l15;
  const __bf16* __restrict__ arow = A + (size_t)row * K;

  v8f acc = {};
#pragma unroll
  for (int kk = 0; kk < K; kk += 32) {
    const int ka = kk + hi * 16;
    // A frag: elements 0..15 -> K=ka..ka+15
    const v8bf alo = *(const v8bf*)(arow + ka);
    const v8bf ahi = *(const v8bf*)(arow + ka + 8);
    const v16bf a = __builtin_shufflevector(alo, ahi,
        0, 1, 2, 3, 4, 5, 6, 7, 8, 9, 10, 11, 12, 13, 14, 15);
    // B frag: one contiguous 32-byte load of this lane's fragment
    const v16bf b = *(const v16bf*)(Bfrag +
        ((size_t)(((kk >> 5) * 4 + wave) * 32 + lane) << 4));
    acc = __builtin_amdgcn_wmma_f32_16x16x32_bf16(false, a, false, b,
                                                  (short)0, acc, false, false);
  }

  const float bn = bias ? bias[n] : 0.f;
#pragma unroll
  for (int j = 0; j < 8; ++j) {
    const int r = blockIdx.x * 16 + hi * 8 + j;   // C layout: 8 rows per half-wave
    float v = acc[j] + bn;
    if (act) v = elu1(v);
    C[(size_t)r * 64 + n] = v;
  }
}

// ---------------------------------------------------------------------------
// Small scalar GEMM for the 16-row device matrices.
// ---------------------------------------------------------------------------
__global__ void small_gemm_kernel(const float* __restrict__ A, const float* __restrict__ W,
                                  const float* __restrict__ bias, float* __restrict__ C,
                                  int M, int K, int N, int act) {
  const int idx = blockIdx.x * blockDim.x + threadIdx.x;
  if (idx >= M * N) return;
  const int m = idx / N, n = idx % N;
  float v = bias ? bias[n] : 0.f;
  for (int k = 0; k < K; ++k) v += A[m * K + k] * W[k * N + n];
  if (act) v = elu1(v);
  C[idx] = v;
}

// ---------------------------------------------------------------------------
// Utility kernels
// ---------------------------------------------------------------------------
__global__ void zero_kernel(float* __restrict__ p, int n) {
  const int i = blockIdx.x * blockDim.x + threadIdx.x;
  if (i < n) p[i] = 0.f;
}

__global__ void count_kernel(const int* __restrict__ dst, int nE, float* __restrict__ deg) {
  const int i = blockIdx.x * blockDim.x + threadIdx.x;
  if (i < nE) atomicAdd(&deg[dst[i]], 1.0f);
}

__global__ void recip_kernel(float* __restrict__ deg, int n) {
  const int i = blockIdx.x * blockDim.x + threadIdx.x;
  if (i < n) deg[i] = 1.0f / fmaxf(deg[i], 1.0f);
}

__global__ void update_kernel(float* __restrict__ x, const float* __restrict__ acc,
                              int n, int act) {
  const int i = blockIdx.x * blockDim.x + threadIdx.x;
  if (i >= n) return;
  float v = x[i] + acc[i];
  if (act) v = elu1(v);
  x[i] = v;
}

// ---------------------------------------------------------------------------
// Per-edge message + mean scatter:
//   msg = T[src] + elu(raw @ W8 + b8) @ We + b ; acc[dst] += msg * rdeg[dst] * scale
// 4 edges per 256-thread block; edge features staged in LDS.
// ---------------------------------------------------------------------------
__global__ void __launch_bounds__(256)
scatter_kernel(const int* __restrict__ src, const int* __restrict__ dst, int nE,
               const float* __restrict__ raw,             // [nE x 8]
               const float* __restrict__ W8, const float* __restrict__ b8,   // edge_trans 8x8
               const float* __restrict__ We,              // 8x64 (rows 64..71 of gconv W)
               const float* __restrict__ bmsg,            // 64
               const float* __restrict__ T,               // [nSrc x 64]
               const float* __restrict__ rdeg,            // [nDst]
               float scale, float* __restrict__ acc) {
  __shared__ float ef[4][8];
  const int t  = threadIdx.x;
  const int e0 = blockIdx.x * 4;
  // stage edge features (post edge_trans elu) for the 4 edges of this block
  if (t < 32) {
    const int ee = e0 + (t >> 3), c = t & 7;
    float v = 0.f;
    if (ee < nE) {
      const float* __restrict__ r = raw + (size_t)ee * 8;
      v = b8[c];
#pragma unroll
      for (int q = 0; q < 8; ++q) v += r[q] * W8[q * 8 + c];
      v = elu1(v);
    }
    ef[t >> 3][c] = v;
  }
  __syncthreads();

  const int le = t >> 6;   // local edge 0..3
  const int j  = t & 63;   // message component
  const int e  = e0 + le;
  if (e >= nE) return;
  const int s = src[e], d = dst[e];
  float m = T[(size_t)s * 64 + j] + bmsg[j];
#pragma unroll
  for (int q = 0; q < 8; ++q) m += ef[le][q] * We[q * 64 + j];
  atomicAdd(&acc[(size_t)d * 64 + j], m * rdeg[d] * scale);
}

// ---------------------------------------------------------------------------
// Final per-place-edge head: concat(op_f[src], edge_f, dev_f[dst]) @ W136x3 + b
// ---------------------------------------------------------------------------
__global__ void final_place_kernel(const int* __restrict__ psrc, const int* __restrict__ pdst,
                                   int nE, const float* __restrict__ place_feats,
                                   const float* __restrict__ W8p, const float* __restrict__ b8p,
                                   const float* __restrict__ opf, const float* __restrict__ devf,
                                   const float* __restrict__ Wf, const float* __restrict__ bf,
                                   float* __restrict__ out) {
  const int e = blockIdx.x * blockDim.x + threadIdx.x;
  if (e >= nE) return;
  float efv[8];
  const float* __restrict__ r = place_feats + (size_t)e * 8;
#pragma unroll
  for (int c = 0; c < 8; ++c) {
    float v = b8p[c];
#pragma unroll
    for (int q = 0; q < 8; ++q) v += r[q] * W8p[q * 8 + c];
    efv[c] = elu1(v);
  }
  const int s = psrc[e], d = pdst[e];
  float o0 = bf[0], o1 = bf[1], o2 = bf[2];
  const float* __restrict__ of = opf + (size_t)s * 64;
  for (int j = 0; j < 64; ++j) {
    const float x = of[j];
    o0 += x * Wf[j * 3 + 0]; o1 += x * Wf[j * 3 + 1]; o2 += x * Wf[j * 3 + 2];
  }
#pragma unroll
  for (int q = 0; q < 8; ++q) {
    const float x = efv[q]; const int rr = 64 + q;
    o0 += x * Wf[rr * 3 + 0]; o1 += x * Wf[rr * 3 + 1]; o2 += x * Wf[rr * 3 + 2];
  }
  const float* __restrict__ df = devf + (size_t)d * 64;
  for (int j = 0; j < 64; ++j) {
    const float x = df[j]; const int rr = 72 + j;
    o0 += x * Wf[rr * 3 + 0]; o1 += x * Wf[rr * 3 + 1]; o2 += x * Wf[rr * 3 + 2];
  }
  out[(size_t)e * 3 + 0] = o0;
  out[(size_t)e * 3 + 1] = o1;
  out[(size_t)e * 3 + 2] = o2;
}

__global__ void nccl_kernel(const float* __restrict__ opf, const float* __restrict__ W,
                            const float* __restrict__ b, float* __restrict__ out) {
  const int n = blockIdx.x * blockDim.x + threadIdx.x;
  if (n >= N_OP) return;
  float v = b[0];
  const float* __restrict__ of = opf + (size_t)n * 64;
  for (int j = 0; j < 64; ++j) v += of[j] * W[j];
  out[n] = v;
}

// ---------------------------------------------------------------------------
// Host side
// ---------------------------------------------------------------------------
// etype internal order: 0=link 1=prev 2=succ 3=place 4=serve
struct Idx {
  int opFeats, devFeats, tensorF, linkF, placeF;
  int linkS, linkD, prevS, prevD, succS, succD, placeS, placeD, serveS, serveD;
  int opW, opB, devW, devB;
  int eW[5], eB[5];
  int gW[6][5], gB[6][5];
  int fpW, fpB, fnW, fnB;
};

static void fill_insertion(Idx& I) {
  I.opFeats = 0; I.devFeats = 1; I.tensorF = 2; I.linkF = 3; I.placeF = 4;
  I.linkS = 5; I.linkD = 6; I.prevS = 7; I.prevD = 8; I.succS = 9; I.succD = 10;
  I.placeS = 11; I.placeD = 12; I.serveS = 13; I.serveD = 14;
  I.opW = 15; I.opB = 16; I.devW = 17; I.devB = 18;
  for (int e = 0; e < 5; ++e) { I.eW[e] = 19 + 2 * e; I.eB[e] = 20 + 2 * e; }
  for (int l = 0; l < 6; ++l)
    for (int e = 0; e < 5; ++e) { I.gW[l][e] = 29 + l * 10 + 2 * e; I.gB[l][e] = I.gW[l][e] + 1; }
  I.fpW = 89; I.fpB = 90; I.fnW = 91; I.fnB = 92;
}

static void fill_sorted(Idx& I) {
  // jax tree_flatten: dict keys sorted alphabetically, recursively
  I.devFeats = 0; I.linkD = 1; I.linkF = 2; I.linkS = 3; I.opFeats = 4;
  I.devW = 5; I.devB = 6;
  // edge_trans sorted: link, place, prev, serve, succ
  I.eW[0] = 7;  I.eB[0] = 8;    // link
  I.eW[3] = 9;  I.eB[3] = 10;   // place
  I.eW[1] = 11; I.eB[1] = 12;   // prev
  I.eW[4] = 13; I.eB[4] = 14;   // serve
  I.eW[2] = 15; I.eB[2] = 16;   // succ
  I.fnW = 17; I.fnB = 18;
  I.fpW = 19; I.fpB = 20;
  for (int l = 0; l < 6; ++l) {
    const int b = 21 + l * 10;
    I.gW[l][0] = b + 0; I.gB[l][0] = b + 1;   // link
    I.gW[l][3] = b + 2; I.gB[l][3] = b + 3;   // place
    I.gW[l][1] = b + 4; I.gB[l][1] = b + 5;   // prev
    I.gW[l][4] = b + 6; I.gB[l][4] = b + 7;   // serve
    I.gW[l][2] = b + 8; I.gB[l][2] = b + 9;   // succ
  }
  I.opW = 81; I.opB = 82;
  I.placeD = 83; I.placeF = 84; I.placeS = 85;
  I.prevD = 86; I.prevS = 87;
  I.serveD = 88; I.serveS = 89;
  I.succD = 90; I.succS = 91;
  I.tensorF = 92;
}

extern "C" void kernel_launch(void* const* d_in, const int* in_sizes, int n_in,
                              void* d_out, int out_size, void* d_ws, size_t ws_size,
                              hipStream_t stream) {
  Idx I;
  if (n_in > 0 && in_sizes[0] == N_OP * 256) fill_insertion(I);
  else fill_sorted(I);

  auto F = [&](int i) -> const float* { return (const float*)d_in[i]; };
  auto Ii = [&](int i) -> const int*  { return (const int*)d_in[i]; };

  // workspace layout (floats, all offsets multiples of 16 -> 64B aligned)
  float* w = (float*)d_ws;
  float* op_f    = w; w += (size_t)N_OP * 64;
  float* T       = w; w += (size_t)N_OP * 64;
  float* op_acc  = w; w += (size_t)N_OP * 64;
  float* dev_f   = w; w += 16 * 64;
  float* dev_T   = w; w += 16 * 64;
  float* dev_acc = w; w += 16 * 64;
  float* rdeg    = w; w += 3 * N_OP + 192;     // contiguous: prev|succ|serve|place|link
  float* rdeg_prev  = rdeg;
  float* rdeg_succ  = rdeg + N_OP;
  float* rdeg_serve = rdeg + 2 * N_OP;
  float* rdeg_place = rdeg + 3 * N_OP;
  float* rdeg_link  = rdeg + 3 * N_OP + 64;
  const int nDeg = 3 * N_OP + 128;
  // bf16 regions (reuse float pointer arithmetic: 1 float = 2 bf16)
  __bf16* opfeats_bf = (__bf16*)w; w += (size_t)N_OP * 128;   // 50000 x 256 bf16
  __bf16* opf_bf     = (__bf16*)w; w += (size_t)N_OP * 32;    // 50000 x 64 bf16
  __bf16* wfrag256   = (__bf16*)w; w += 8192;                 // 16384 bf16
  __bf16* wfrag64    = (__bf16*)w; w += 2048;                 // 4096 bf16

  const dim3 b256(256);
  auto g1 = [](int n) { return dim3((unsigned)((n + 255) / 256)); };

  // ---- in-degrees (recomputed each call; cheap, capture-safe) ----
  zero_kernel<<<g1(nDeg), b256, 0, stream>>>(rdeg, nDeg);
  count_kernel<<<g1(E_PP), b256, 0, stream>>>(Ii(I.prevD),  E_PP, rdeg_prev);
  count_kernel<<<g1(E_PP), b256, 0, stream>>>(Ii(I.succD),  E_PP, rdeg_succ);
  count_kernel<<<g1(E_PD), b256, 0, stream>>>(Ii(I.serveD), E_PD, rdeg_serve);
  count_kernel<<<g1(E_PD), b256, 0, stream>>>(Ii(I.placeD), E_PD, rdeg_place);
  count_kernel<<<g1(E_DD), b256, 0, stream>>>(Ii(I.linkD),  E_DD, rdeg_link);
  recip_kernel<<<g1(nDeg), b256, 0, stream>>>(rdeg, nDeg);

  // ---- input transforms ----
  cvt_bf16_kernel<<<g1(N_OP * 128), b256, 0, stream>>>(F(I.opFeats), opfeats_bf, N_OP * 128);
  pack_w_kernel<<<g1(16384), b256, 0, stream>>>(F(I.opW), wfrag256, 256);
  gemm64_bf_kernel<256><<<dim3(N_OP / 16), dim3(128), 0, stream>>>(
      opfeats_bf, wfrag256, F(I.opB), op_f, 1);
  small_gemm_kernel<<<g1(16 * 64), b256, 0, stream>>>(
      F(I.devFeats), F(I.devW), F(I.devB), dev_f, 16, 16, 64, 1);

  // ---- 6 hetero graph-conv layers ----
  const float third = 1.0f / 3.0f, half = 0.5f;
  for (int l = 0; l < 6; ++l) {
    const int act = (l < 5) ? 1 : 0;
    zero_kernel<<<g1(N_OP * 64), b256, 0, stream>>>(op_acc, N_OP * 64);
    zero_kernel<<<g1(16 * 64), b256, 0, stream>>>(dev_acc, 16 * 64);

    // bf16 snapshot of current op features (A operand for the three T GEMMs)
    cvt_bf16_kernel<<<g1(N_OP * 32), b256, 0, stream>>>(op_f, opf_bf, N_OP * 32);

    // prev : op -> op
    pack_w_kernel<<<g1(4096), b256, 0, stream>>>(F(I.gW[l][1]), wfrag64, 64);
    gemm64_bf_kernel<64><<<dim3(N_OP / 16), dim3(128), 0, stream>>>(
        opf_bf, wfrag64, nullptr, T, 0);
    scatter_kernel<<<dim3((E_PP + 3) / 4), b256, 0, stream>>>(
        Ii(I.prevS), Ii(I.prevD), E_PP, F(I.tensorF), F(I.eW[1]), F(I.eB[1]),
        F(I.gW[l][1]) + 64 * 64, F(I.gB[l][1]), T, rdeg_prev, third, op_acc);

    // succ : op -> op
    pack_w_kernel<<<g1(4096), b256, 0, stream>>>(F(I.gW[l][2]), wfrag64, 64);
    gemm64_bf_kernel<64><<<dim3(N_OP / 16), dim3(128), 0, stream>>>(
        opf_bf, wfrag64, nullptr, T, 0);
    scatter_kernel<<<dim3((E_PP + 3) / 4), b256, 0, stream>>>(
        Ii(I.succS), Ii(I.succD), E_PP, F(I.tensorF), F(I.eW[2]), F(I.eB[2]),
        F(I.gW[l][2]) + 64 * 64, F(I.gB[l][2]), T, rdeg_succ, third, op_acc);

    // place : op -> device
    pack_w_kernel<<<g1(4096), b256, 0, stream>>>(F(I.gW[l][3]), wfrag64, 64);
    gemm64_bf_kernel<64><<<dim3(N_OP / 16), dim3(128), 0, stream>>>(
        opf_bf, wfrag64, nullptr, T, 0);
    scatter_kernel<<<dim3((E_PD + 3) / 4), b256, 0, stream>>>(
        Ii(I.placeS), Ii(I.placeD), E_PD, F(I.placeF), F(I.eW[3]), F(I.eB[3]),
        F(I.gW[l][3]) + 64 * 64, F(I.gB[l][3]), T, rdeg_place, half, dev_acc);

    // serve : device -> op
    small_gemm_kernel<<<g1(16 * 64), b256, 0, stream>>>(
        dev_f, F(I.gW[l][4]), nullptr, dev_T, 16, 64, 64, 0);
    scatter_kernel<<<dim3((E_PD + 3) / 4), b256, 0, stream>>>(
        Ii(I.serveS), Ii(I.serveD), E_PD, F(I.placeF), F(I.eW[4]), F(I.eB[4]),
        F(I.gW[l][4]) + 64 * 64, F(I.gB[l][4]), dev_T, rdeg_serve, third, op_acc);

    // link : device -> device
    small_gemm_kernel<<<g1(16 * 64), b256, 0, stream>>>(
        dev_f, F(I.gW[l][0]), nullptr, dev_T, 16, 64, 64, 0);
    scatter_kernel<<<dim3((E_DD + 3) / 4), b256, 0, stream>>>(
        Ii(I.linkS), Ii(I.linkD), E_DD, F(I.linkF), F(I.eW[0]), F(I.eB[0]),
        F(I.gW[l][0]) + 64 * 64, F(I.gB[l][0]), dev_T, rdeg_link, half, dev_acc);

    // residual update + activation
    update_kernel<<<g1(N_OP * 64), b256, 0, stream>>>(op_f, op_acc, N_OP * 64, act);
    update_kernel<<<g1(16 * 64), b256, 0, stream>>>(dev_f, dev_acc, 16 * 64, act);
  }

  // ---- heads ----
  float* out = (float*)d_out;
  final_place_kernel<<<g1(E_PD), b256, 0, stream>>>(
      Ii(I.placeS), Ii(I.placeD), E_PD, F(I.placeF), F(I.eW[3]), F(I.eB[3]),
      op_f, dev_f, F(I.fpW), F(I.fpB), out);
  nccl_kernel<<<g1(N_OP), b256, 0, stream>>>(
      op_f, F(I.fnW), F(I.fnB), out + (size_t)E_PD * 3);
}
